// Qwen3MoeSparseMoeBlock_9053791060131
// MI455X (gfx1250) — matrix operand order repaired and checked by reading, not compiled, not run on hardware
//
#include <hip/hip_runtime.h>
#include <hip/hip_bf16.h>
#include <stdint.h>

// Problem sizes (fixed by the reference)
constexpr int NT = 4096;   // tokens
constexpr int ND = 2048;   // hidden
constexpr int NE = 64;     // experts
constexpr int NK = 8;      // top-k
constexpr int NF = 768;    // intermediate
constexpr int NC = 1024;   // capacity per expert

constexpr int KC = 64;     // K chunk staged in LDS
constexpr int LP = KC + 8; // padded LDS row stride (elements) -> bank spread

typedef __attribute__((ext_vector_type(16))) __bf16 bf16x16;
typedef __attribute__((ext_vector_type(8)))  __bf16 bf16x8;
typedef __attribute__((ext_vector_type(8)))  float  f32x8;

#define WMMA_BF16(a, b, c) \
  __builtin_amdgcn_wmma_f32_16x16x32_bf16(false, (a), false, (b), (short)0, (c), false, false)

// ---------------------------------------------------------------------------
// Kernel 0: init output accumulator + slot tables + expert counters
// ---------------------------------------------------------------------------
__global__ void moe_init_kernel(float* __restrict__ y,
                                int* __restrict__ slot_token,
                                float* __restrict__ slot_wt,
                                int* __restrict__ counts) {
  int i = blockIdx.x * blockDim.x + threadIdx.x;
  if (i < NT * ND) y[i] = 0.0f;
  if (i < NE * NC) { slot_token[i] = -1; slot_wt[i] = 0.0f; }
  if (i < NE) counts[i] = 0;
}

// ---------------------------------------------------------------------------
// Kernel 1: router — logits, softmax, top-8, renormalize, slot assignment
// ---------------------------------------------------------------------------
__global__ void moe_router_kernel(const float* __restrict__ hs,
                                  const float* __restrict__ gw,
                                  int* __restrict__ slot_token,
                                  float* __restrict__ slot_wt,
                                  int* __restrict__ counts) {
  __shared__ float row[ND];
  __shared__ float logits[NE];
  __shared__ float probs[NE];

  const int t = blockIdx.x;
  for (int i = threadIdx.x; i < ND; i += blockDim.x)
    row[i] = hs[(size_t)t * ND + i];
  __syncthreads();

  {
    const int e = threadIdx.x;                  // 64 threads == 64 experts
    const float* w = gw + (size_t)e * ND;
    float acc = 0.0f;
    for (int i = 0; i < ND; i += 4) {
      acc += row[i + 0] * w[i + 0] + row[i + 1] * w[i + 1]
           + row[i + 2] * w[i + 2] + row[i + 3] * w[i + 3];
    }
    logits[e] = acc;
  }
  __syncthreads();

  if (threadIdx.x == 0) {
    float m = logits[0];
    for (int i = 1; i < NE; i++) m = fmaxf(m, logits[i]);
    float s = 0.0f;
    for (int i = 0; i < NE; i++) { float p = __expf(logits[i] - m); probs[i] = p; s += p; }
    const float inv_s = __builtin_amdgcn_rcpf(s);

    int   ids[NK];
    float ws[NK];
    float wsum = 0.0f;
    for (int k = 0; k < NK; k++) {
      int bi = 0; float bv = -1.0f;
      for (int i = 0; i < NE; i++) if (probs[i] > bv) { bv = probs[i]; bi = i; }
      probs[bi] = -2.0f;
      ids[k] = bi;
      ws[k]  = bv * inv_s;
      wsum  += ws[k];
    }
    const float inv_w = __builtin_amdgcn_rcpf(wsum);
    for (int k = 0; k < NK; k++) {
      const int eid = ids[k];
      const int pos = atomicAdd(&counts[eid], 1);
      if (pos < NC) {
        slot_token[eid * NC + pos] = t;
        slot_wt[eid * NC + pos]    = ws[k] * inv_w;
      }
    }
  }
}

// ---------------------------------------------------------------------------
// WMMA fragment loaders from LDS tiles (wave32 layouts, ISA 7.12.2).
// Tiles are stored [row][k] bf16 with padded stride LP.
// A frag (16x32 MxK): lanes 0-15: M=lane, K {s..s+7, s+16..s+23};
//                     lanes 16-31: M=lane-16, K {s+8..s+15, s+24..s+31}.
// B frag (32x16 KxN): lane n (0-15): col n, K {s..s+7, s+16..s+23}; lane n+16: K {s+8..s+15, s+24..s+31}.
// ---------------------------------------------------------------------------
__device__ static inline bf16x16 frag_a(const __bf16* base, int row0, int s, int lane) {
  const __bf16* p = base + (row0 + (lane & 15)) * LP + s + ((lane >> 4) << 3);
  union { bf16x16 v; bf16x8 h[2]; } u;
  u.h[0] = *(const bf16x8*)(p);
  u.h[1] = *(const bf16x8*)(p + 16);
  return u.v;
}

__device__ static inline bf16x16 frag_b(const __bf16* base, int row0, int s, int lane) {
  const __bf16* p = base + (row0 + (lane & 15)) * LP + s + ((lane >> 4) << 3);
  union { bf16x16 v; bf16x8 h[2]; } u;
  u.h[0] = *(const bf16x8*)(p);
  u.h[1] = *(const bf16x8*)(p + 16);
  return u.v;
}

__device__ static inline void cvt_store8(__bf16* dst, const float4 a, const float4 b) {
  dst[0] = (__bf16)a.x; dst[1] = (__bf16)a.y; dst[2] = (__bf16)a.z; dst[3] = (__bf16)a.w;
  dst[4] = (__bf16)b.x; dst[5] = (__bf16)b.y; dst[6] = (__bf16)b.z; dst[7] = (__bf16)b.w;
}

// ---------------------------------------------------------------------------
// Kernel 2 (stage 1): act[e, m, f] = silu(A@w1^T) * (A@w3^T), bf16 out.
// Block tile: 128 slots x 64 f-cols; B LDS holds w1 panel (rows 0-63) and
// w3 panel (rows 64-127) so the gate is block-local. 8 waves, 8 accums each.
// ---------------------------------------------------------------------------
__launch_bounds__(256)
__global__ void moe_stage1_kernel(const float* __restrict__ hs,
                                  const float* __restrict__ w1,
                                  const float* __restrict__ w3,
                                  const int* __restrict__ slot_token,
                                  __bf16* __restrict__ act) {
  __shared__ __align__(16) __bf16 At[128 * LP];
  __shared__ __align__(16) __bf16 Bt[128 * LP];
  __shared__ int stokL[128];

  const int e  = blockIdx.z;
  const int m0 = blockIdx.y * 128;
  const int f0 = blockIdx.x * 64;

  if (slot_token[e * NC + m0] < 0) return;   // capacity tile fully empty (uniform)

  const int tid = threadIdx.x;
  const int wave = tid >> 5;
  const int lane = tid & 31;

  if (tid < 128) stokL[tid] = slot_token[e * NC + m0 + tid];
  __syncthreads();

  // per-thread staging assignment: 2 threads per row, 32 K-elems each
  const int r    = tid >> 1;
  const int half = tid & 1;
  const int tokr = stokL[r];
  const float* asrc = hs + (size_t)(tokr < 0 ? 0 : tokr) * ND + half * 32;
  const float* bsrc = (r < 64)
      ? (w1 + ((size_t)e * NF + f0 + r) * ND + half * 32)
      : (w3 + ((size_t)e * NF + f0 + (r - 64)) * ND + half * 32);
  __bf16* adst = &At[r * LP + half * 32];
  __bf16* bdst = &Bt[r * LP + half * 32];

  const int mtp = wave & 3;      // M-tile pair: {2*mtp, 2*mtp+1} of 8
  const int ftp = wave >> 2;     // F-tile pair: {2*ftp, 2*ftp+1} of 4

  f32x8 acc1[2][2] = {};
  f32x8 acc3[2][2] = {};

  for (int kc = 0; kc < ND; kc += KC) {
    __syncthreads();                         // previous chunk consumed
#pragma unroll
    for (int i = 0; i < 32; i += 8) {
      float4 a0 = make_float4(0.f, 0.f, 0.f, 0.f), a1 = a0;
      if (tokr >= 0) { a0 = *(const float4*)(asrc + kc + i);
                       a1 = *(const float4*)(asrc + kc + i + 4); }
      const float4 b0 = *(const float4*)(bsrc + kc + i);
      const float4 b1 = *(const float4*)(bsrc + kc + i + 4);
      cvt_store8(adst + i, a0, a1);
      cvt_store8(bdst + i, b0, b1);
    }
    __syncthreads();                         // chunk staged

#pragma unroll
    for (int s = 0; s < KC; s += 32) {
      const bf16x16 a0 = frag_a(At, (2 * mtp + 0) * 16, s, lane);
      const bf16x16 a1 = frag_a(At, (2 * mtp + 1) * 16, s, lane);
      const bf16x16 p0 = frag_b(Bt, (2 * ftp + 0) * 16, s, lane);        // w1 panel
      const bf16x16 p1 = frag_b(Bt, (2 * ftp + 1) * 16, s, lane);
      const bf16x16 q0 = frag_b(Bt, 64 + (2 * ftp + 0) * 16, s, lane);   // w3 panel
      const bf16x16 q1 = frag_b(Bt, 64 + (2 * ftp + 1) * 16, s, lane);
      acc1[0][0] = WMMA_BF16(a0, p0, acc1[0][0]);
      acc1[0][1] = WMMA_BF16(a0, p1, acc1[0][1]);
      acc1[1][0] = WMMA_BF16(a1, p0, acc1[1][0]);
      acc1[1][1] = WMMA_BF16(a1, p1, acc1[1][1]);
      acc3[0][0] = WMMA_BF16(a0, q0, acc3[0][0]);
      acc3[0][1] = WMMA_BF16(a0, q1, acc3[0][1]);
      acc3[1][0] = WMMA_BF16(a1, q0, acc3[1][0]);
      acc3[1][1] = WMMA_BF16(a1, q1, acc3[1][1]);
    }
  }

  // epilogue: act = silu(h1) * h3  (fast rcp-based sigmoid), bf16 store
  const int n  = lane & 15;
  const int rb = (lane >> 4) << 3;           // C/D layout: VGPR r -> M = r + rb
#pragma unroll
  for (int i = 0; i < 2; i++) {
#pragma unroll
    for (int j = 0; j < 2; j++) {
      const int mg = m0 + (2 * mtp + i) * 16 + rb;
      const int fg = f0 + (2 * ftp + j) * 16 + n;
#pragma unroll
      for (int rr = 0; rr < 8; rr++) {
        const float g   = acc1[i][j][rr];
        const float sig = __builtin_amdgcn_rcpf(1.0f + __expf(-g));
        act[(size_t)(e * NC + mg + rr) * NF + fg] = (__bf16)(g * sig * acc3[i][j][rr]);
      }
    }
  }
}

// ---------------------------------------------------------------------------
// Kernel 3 (stage 2): out = act @ w2^T, weighted atomic scatter into y.
// Block tile: 128 slots x 128 d-cols. act is already bf16 -> A chunks are
// fetched with the CDNA5 async-to-LDS path (ASYNCcnt), no VGPR round-trip.
// ---------------------------------------------------------------------------
__launch_bounds__(256)
__global__ void moe_stage2_kernel(const __bf16* __restrict__ act,
                                  const float* __restrict__ w2,
                                  const int* __restrict__ slot_token,
                                  const float* __restrict__ slot_wt,
                                  float* __restrict__ y) {
  __shared__ __align__(16) __bf16 At[128 * LP];
  __shared__ __align__(16) __bf16 Bt[128 * LP];
  __shared__ int   stokL[128];
  __shared__ float swtL[128];

  const int e  = blockIdx.z;
  const int m0 = blockIdx.y * 128;
  const int d0 = blockIdx.x * 128;

  if (slot_token[e * NC + m0] < 0) return;   // uniform early exit

  const int tid = threadIdx.x;
  const int wave = tid >> 5;
  const int lane = tid & 31;

  if (tid < 128) {
    stokL[tid] = slot_token[e * NC + m0 + tid];
    swtL[tid]  = slot_wt[e * NC + m0 + tid];
  }

  const int r    = tid >> 1;
  const int half = tid & 1;
  const float* bsrc = w2 + ((size_t)e * ND + d0 + r) * NF + half * 32;
  __bf16* bdst = &Bt[r * LP + half * 32];

  const int mtp = wave & 3;      // M-tiles {2*mtp, 2*mtp+1}
  const int ntp = wave >> 2;     // N-tiles {4*ntp .. 4*ntp+3}

  f32x8 acc[2][4] = {};

  for (int kc = 0; kc < NF; kc += KC) {
    __syncthreads();

    // A chunk: 128 rows x 64 bf16 == 1024 x 16B, async global -> LDS
#pragma unroll
    for (int i = 0; i < 4; i++) {
      const int idx = tid + i * 256;
      const int row = idx >> 3;
      const int c8  = idx & 7;
      const unsigned ldso = (unsigned)(uintptr_t)&At[row * LP + c8 * 8];
      const __bf16* g = act + (size_t)(e * NC + m0 + row) * NF + kc + c8 * 8;
      asm volatile("global_load_async_to_lds_b128 %0, %1, off"
                   :: "v"(ldso), "v"(g) : "memory");
    }

    // B chunk: w2 panel fp32 -> bf16 through VGPRs
#pragma unroll
    for (int i = 0; i < 32; i += 8) {
      const float4 b0 = *(const float4*)(bsrc + kc + i);
      const float4 b1 = *(const float4*)(bsrc + kc + i + 4);
      cvt_store8(bdst + i, b0, b1);
    }

    asm volatile("s_wait_asynccnt 0x0" ::: "memory");   // my async copies done
    __syncthreads();                                    // everyone's staging done

#pragma unroll
    for (int s = 0; s < KC; s += 32) {
      const bf16x16 a0 = frag_a(At, (2 * mtp + 0) * 16, s, lane);
      const bf16x16 a1 = frag_a(At, (2 * mtp + 1) * 16, s, lane);
#pragma unroll
      for (int j = 0; j < 4; j++) {
        const bf16x16 b = frag_b(Bt, (4 * ntp + j) * 16, s, lane);
        acc[0][j] = WMMA_BF16(a0, b, acc[0][j]);
        acc[1][j] = WMMA_BF16(a1, b, acc[1][j]);
      }
    }
  }

  // epilogue: weighted combine via global f32 atomics
  const int n  = lane & 15;
  const int rb = (lane >> 4) << 3;
#pragma unroll
  for (int i = 0; i < 2; i++) {
    const int ml = (2 * mtp + i) * 16 + rb;
#pragma unroll
    for (int j = 0; j < 4; j++) {
      const int dg = d0 + (4 * ntp + j) * 16 + n;
#pragma unroll
      for (int rr = 0; rr < 8; rr++) {
        const int tok = stokL[ml + rr];
        if (tok >= 0)
          atomicAdd(&y[(size_t)tok * ND + dg], acc[i][j][rr] * swtL[ml + rr]);
      }
    }
  }
}

// ---------------------------------------------------------------------------
// Launch
// ---------------------------------------------------------------------------
extern "C" void kernel_launch(void* const* d_in, const int* in_sizes, int n_in,
                              void* d_out, int out_size, void* d_ws, size_t ws_size,
                              hipStream_t stream) {
  const float* hs = (const float*)d_in[0];   // [T, D]
  const float* gw = (const float*)d_in[1];   // [E, D]
  const float* w1 = (const float*)d_in[2];   // [E, F, D]
  const float* w3 = (const float*)d_in[3];   // [E, F, D]
  const float* w2 = (const float*)d_in[4];   // [E, D, F]
  float* y = (float*)d_out;                  // [T, D]

  char* ws = (char*)d_ws;
  int*    counts     = (int*)ws;                                    // [E]
  int*    slot_token = (int*)(ws + 256);                            // [E*C]
  float*  slot_wt    = (float*)(ws + 256 + sizeof(int) * NE * NC);  // [E*C]
  __bf16* act        = (__bf16*)(ws + 256 + 2u * sizeof(int) * NE * NC); // [E*C*F] bf16

  moe_init_kernel<<<(NT * ND + 255) / 256, 256, 0, stream>>>(y, slot_token, slot_wt, counts);
  moe_router_kernel<<<NT, 64, 0, stream>>>(hs, gw, slot_token, slot_wt, counts);

  dim3 g1(NF / 64, NC / 128, NE);
  moe_stage1_kernel<<<g1, 256, 0, stream>>>(hs, w1, w3, slot_token, act);

  dim3 g2(ND / 128, NC / 128, NE);
  moe_stage2_kernel<<<g2, 256, 0, stream>>>(act, w2, slot_token, slot_wt, y);
}
